// ExpertTransformer_8280696947248
// MI455X (gfx1250) — matrix operand order repaired and checked by reading, not compiled, not run on hardware
//
#include <hip/hip_runtime.h>

// ---------------------------------------------------------------------------
// Model constants (from reference)
// ---------------------------------------------------------------------------
enum : int { B_ = 16, T_ = 256, KW_ = 64, H_ = 1024, NH_ = 16, HD_ = 64,
             L_ = 6, V_ = 32000, E_ = 2048, VT_ = 49, F_ = 512 };

static constexpr int BM = 128, BN = 64, BK = 32;
static constexpr int LKU = BK / 2 + 1;   // uint row stride 17 -> 68B, odd bank step

typedef __attribute__((ext_vector_type(16))) __bf16 v16bf;
typedef __attribute__((ext_vector_type(2)))  __bf16 v2bf;
typedef __attribute__((ext_vector_type(2)))  float  v2f;
typedef __attribute__((ext_vector_type(8)))  float  v8f;

// Packed bf16 conversion (RTNE): fptrunc <2 x float> -> <2 x bfloat>
// selects a single v_cvt_pk_bf16_f32.
__device__ __forceinline__ unsigned pack2(float lo, float hi) {
    v2f f; f.x = lo; f.y = hi;
    v2bf h = __builtin_convertvector(f, v2bf);
    union { v2bf h; unsigned u; } c; c.h = h;
    return c.u;
}
__device__ __forceinline__ float gelu_erf(float v) {
    return 0.5f * v * (1.0f + erff(v * 0.70710678118654752f));
}
__device__ __forceinline__ int clampi(int v, int hi) { return v < hi ? v : hi; }

// ---------------------------------------------------------------------------
// Tile staging: global f32 -> registers -> packed bf16 LDS.
// Slow paths use clamped addresses + select-0 (no divergent branches).
// ---------------------------------------------------------------------------
__device__ __forceinline__ void loadA_tile(const float* __restrict__ A, int row0, int k0,
                                           int M, int K, int lda, bool fast, int tid,
                                           float r[16])
{
    if (fast) {
        #pragma unroll
        for (int t = 0; t < 4; ++t) {                 // 128 rows x 8 quads
            int i = tid + t * 256;
            int rr = i >> 3, q = i & 7;
            const float4 v = *(const float4*)(A + (long)(row0 + rr) * lda + k0 + q * 4);
            r[t*4+0] = v.x; r[t*4+1] = v.y; r[t*4+2] = v.z; r[t*4+3] = v.w;
        }
    } else {
        #pragma unroll
        for (int t = 0; t < 8; ++t) {                 // 128 rows x 16 pairs
            int i = tid + t * 256;
            int rr = i >> 4, kp = i & 15;
            int gr = row0 + rr, gk = k0 + kp * 2;
            long base = (long)clampi(gr, M - 1) * lda;
            float e0 = A[base + clampi(gk,     K - 1)];
            float e1 = A[base + clampi(gk + 1, K - 1)];
            bool rok = gr < M;
            r[t*2]   = (rok && gk     < K) ? e0 : 0.f;
            r[t*2+1] = (rok && gk + 1 < K) ? e1 : 0.f;
        }
    }
}

__device__ __forceinline__ void storeA_tile(unsigned (*Als)[LKU], bool fast, int tid,
                                            const float r[16])
{
    if (fast) {
        #pragma unroll
        for (int t = 0; t < 4; ++t) {
            int i = tid + t * 256;
            int rr = i >> 3, q = i & 7;
            Als[rr][q*2]     = pack2(r[t*4+0], r[t*4+1]);
            Als[rr][q*2 + 1] = pack2(r[t*4+2], r[t*4+3]);
        }
    } else {
        #pragma unroll
        for (int t = 0; t < 8; ++t) {
            int i = tid + t * 256;
            int rr = i >> 4, kp = i & 15;
            Als[rr][kp] = pack2(r[t*2], r[t*2+1]);
        }
    }
}

__device__ __forceinline__ void loadB_tile(const float* __restrict__ Bm, int col0, int k0,
                                           int N, int K, int ldb, bool fast, bool trans,
                                           int tid, float r[8])
{
    if (trans) {                                      // B stored [N,K]
        if (fast) {
            #pragma unroll
            for (int t = 0; t < 2; ++t) {             // 64 rows x 8 quads
                int i = tid + t * 256;
                int n = i >> 3, q = i & 7;
                const float4 v = *(const float4*)(Bm + (long)(col0 + n) * ldb + k0 + q * 4);
                r[t*4+0] = v.x; r[t*4+1] = v.y; r[t*4+2] = v.z; r[t*4+3] = v.w;
            }
        } else {
            #pragma unroll
            for (int t = 0; t < 4; ++t) {             // 64 rows x 16 pairs
                int i = tid + t * 256;
                int n = i >> 4, kp = i & 15;
                int gn = col0 + n, gk = k0 + kp * 2;
                long base = (long)clampi(gn, N - 1) * ldb;
                float e0 = Bm[base + clampi(gk,     K - 1)];
                float e1 = Bm[base + clampi(gk + 1, K - 1)];
                bool nok = gn < N;
                r[t*2]   = (nok && gk     < K) ? e0 : 0.f;
                r[t*2+1] = (nok && gk + 1 < K) ? e1 : 0.f;
            }
        }
    } else {                                          // B stored [K,N]
        if (fast) {
            int nq = tid & 15, kp = tid >> 4;         // 16 n-quads x 16 k-pairs
            int gk = k0 + kp * 2, gn = col0 + nq * 4;
            const float4 r0 = *(const float4*)(Bm + (long)gk * ldb + gn);
            const float4 r1 = *(const float4*)(Bm + (long)(gk + 1) * ldb + gn);
            r[0] = r0.x; r[1] = r0.y; r[2] = r0.z; r[3] = r0.w;
            r[4] = r1.x; r[5] = r1.y; r[6] = r1.z; r[7] = r1.w;
        } else {
            #pragma unroll
            for (int t = 0; t < 4; ++t) {             // pairs along k, coalesced in n
                int i = tid + t * 256;
                int n = i & 63, kp = i >> 6;
                int gn = col0 + n, gk = k0 + kp * 2;
                int cn = clampi(gn, N - 1);
                float e0 = Bm[(long)clampi(gk,     K - 1) * ldb + cn];
                float e1 = Bm[(long)clampi(gk + 1, K - 1) * ldb + cn];
                bool nok = gn < N;
                r[t*2]   = (nok && gk     < K) ? e0 : 0.f;
                r[t*2+1] = (nok && gk + 1 < K) ? e1 : 0.f;
            }
        }
    }
}

__device__ __forceinline__ void storeB_tile(unsigned (*Bls)[LKU], bool fast, bool trans,
                                            int tid, const float r[8])
{
    if (trans) {
        if (fast) {
            #pragma unroll
            for (int t = 0; t < 2; ++t) {
                int i = tid + t * 256;
                int n = i >> 3, q = i & 7;
                Bls[n][q*2]     = pack2(r[t*4+0], r[t*4+1]);
                Bls[n][q*2 + 1] = pack2(r[t*4+2], r[t*4+3]);
            }
        } else {
            #pragma unroll
            for (int t = 0; t < 4; ++t) {
                int i = tid + t * 256;
                int n = i >> 4, kp = i & 15;
                Bls[n][kp] = pack2(r[t*2], r[t*2+1]);
            }
        }
    } else {
        if (fast) {
            int nq = tid & 15, kp = tid >> 4;
            #pragma unroll
            for (int j = 0; j < 4; ++j)
                Bls[nq*4 + j][kp] = pack2(r[j], r[4 + j]);
        } else {
            #pragma unroll
            for (int t = 0; t < 4; ++t) {
                int i = tid + t * 256;
                int n = i & 63, kp = i >> 6;
                Bls[n][kp] = pack2(r[t*2], r[t*2+1]);
            }
        }
    }
}

// ---------------------------------------------------------------------------
// Generic batched strided GEMM:  C = alpha * A @ B (+bias)(+gelu)(+res)
// flags: 1 = B stored [N,K] (compute A.B^T), 2 = +bias[n], 4 = +res (C layout),
//        8 = exact-erf GELU after bias
// Block tile 128x64x32; 8 waves (4m x 2n), each wave owns 32x32 of C
// (2 A-frags x 2 B-frags -> 4 v_wmma per k-chunk). LDS double-buffered:
// one barrier per k-chunk, next tile's global loads overlap the WMMAs.
// ---------------------------------------------------------------------------
__global__ __launch_bounds__(256)
void gemm_bf16_wmma(const float* __restrict__ A, const float* __restrict__ Bm,
                    const float* __restrict__ bias, const float* __restrict__ res,
                    float* __restrict__ C,
                    int M, int N, int K, int lda, int ldb, int ldc, int nbh,
                    long sAb, long sAh, long sBb, long sBh, long sCb, long sCh,
                    float alpha, int flags)
{
    __shared__ __align__(16) unsigned AlsU[2][BM][LKU];
    __shared__ __align__(16) unsigned BlsU[2][BN][LKU];

    const int zb = blockIdx.z / nbh, zh = blockIdx.z % nbh;
    A  += (long)zb * sAb + (long)zh * sAh;
    Bm += (long)zb * sBb + (long)zh * sBh;
    const long coff = (long)zb * sCb + (long)zh * sCh;
    C += coff;
    if (res) res += coff;

    const int row0 = blockIdx.y * BM, col0 = blockIdx.x * BN;
    const int tid  = threadIdx.x;
    const int wave = tid >> 5, lane = tid & 31;
    const int wm = wave >> 1, wn = wave & 1;          // 4 x 2 wave grid
    const int lm = lane & 15, khalf = lane >> 4;

    const bool kfull = (K % BK) == 0;
    const bool aFast = kfull && (row0 + BM <= M) && ((lda & 3) == 0);
    const bool bFast = kfull && (col0 + BN <= N) && ((ldb & 3) == 0);
    const bool trans = (flags & 1) != 0;

    v8f acc00 = {0.f,0.f,0.f,0.f,0.f,0.f,0.f,0.f};
    v8f acc01 = acc00, acc10 = acc00, acc11 = acc00;

    float ar[16], br[8];
    loadA_tile(A, row0, 0, M, K, lda, aFast, tid, ar);
    loadB_tile(Bm, col0, 0, N, K, ldb, bFast, trans, tid, br);
    storeA_tile(AlsU[0], aFast, tid, ar);
    storeB_tile(BlsU[0], bFast, trans, tid, br);
    int cur = 0;

    for (int k0 = 0; k0 < K; k0 += BK) {
        const bool more = (k0 + BK) < K;
        __syncthreads();
        if (more) {                                   // prefetch next stage
            loadA_tile(A, row0, k0 + BK, M, K, lda, aFast, tid, ar);
            loadB_tile(Bm, col0, k0 + BK, N, K, ldb, bFast, trans, tid, br);
        }

        // -------- fragments per documented 16-bit WMMA lane/VGPR layouts ----
        union Frag { v16bf v; unsigned u[8]; };
        Frag a0, a1, b0, b1;
        #pragma unroll
        for (int j = 0; j < 8; ++j) {
            // A 16x32: lanes0-15 V0..3->K0..7, V4..7->K16..23; lanes16-31 +8
            int ka = ((j < 4) ? j : 8 + (j - 4)) + 4 * khalf;     // uint index
            a0.u[j] = AlsU[cur][wm * 32 + lm][ka];
            a1.u[j] = AlsU[cur][wm * 32 + 16 + lm][ka];
            // B 32x16: lanes0-15 V0..3->K0..7, V4..7->K16..23; lanes16-31 +8
            int kb = ((j < 4) ? j : 8 + (j - 4)) + 4 * khalf;     // uint index
            b0.u[j] = BlsU[cur][wn * 32 + lm][kb];
            b1.u[j] = BlsU[cur][wn * 32 + 16 + lm][kb];
        }
        acc00 = __builtin_amdgcn_wmma_f32_16x16x32_bf16(false, a0.v, false, b0.v,
                                                        (short)0, acc00, false, false);
        acc01 = __builtin_amdgcn_wmma_f32_16x16x32_bf16(false, a0.v, false, b1.v,
                                                        (short)0, acc01, false, false);
        acc10 = __builtin_amdgcn_wmma_f32_16x16x32_bf16(false, a1.v, false, b0.v,
                                                        (short)0, acc10, false, false);
        acc11 = __builtin_amdgcn_wmma_f32_16x16x32_bf16(false, a1.v, false, b1.v,
                                                        (short)0, acc11, false, false);
        if (more) {                                   // fill the other buffer
            storeA_tile(AlsU[cur ^ 1], aFast, tid, ar);
            storeB_tile(BlsU[cur ^ 1], bFast, trans, tid, br);
            cur ^= 1;
        }
    }

    // ---- epilogue: C/D layout lane=(n | mh<<4), vgpr r -> m = r + 8*mh ----
    const int nn = lane & 15, mh = lane >> 4;
    #pragma unroll
    for (int sub = 0; sub < 2; ++sub) {
        #pragma unroll
        for (int r = 0; r < 8; ++r) {
            int gm = row0 + wm * 32 + sub * 16 + r + 8 * mh;
            if (gm >= M) continue;
            #pragma unroll
            for (int half = 0; half < 2; ++half) {
                int gn = col0 + wn * 32 + half * 16 + nn;
                if (gn >= N) continue;
                float v;
                if      (sub == 0 && half == 0) v = acc00[r];
                else if (sub == 0)              v = acc01[r];
                else if (half == 0)             v = acc10[r];
                else                            v = acc11[r];
                v *= alpha;
                if (flags & 2) v += bias[gn];
                if (flags & 8) v = gelu_erf(v);
                if (flags & 4) v += res[(long)gm * ldc + gn];
                C[(long)gm * ldc + gn] = v;
            }
        }
    }
}

// ---------------------------------------------------------------------------
// Embedding gather: out[row] = We[tok[row]] (+ wpe[row % T])
// ---------------------------------------------------------------------------
__global__ void embed_kernel(const int* __restrict__ tok, const float* __restrict__ We,
                             const float* __restrict__ wpe, float* __restrict__ out,
                             int T, int Hc)
{
    long row = blockIdx.x;
    int t = (int)(row % T);
    const float* src = We + (long)tok[row] * Hc;
    const float* pe  = wpe ? (wpe + (long)t * Hc) : nullptr;
    for (int j = threadIdx.x; j < Hc; j += 256)
        out[row * (long)Hc + j] = src[j] + (pe ? pe[j] : 0.f);
}

// ---------------------------------------------------------------------------
// Row softmax over att[..., Tq, Tn]; causal => cols <= row valid; masked -> 0
// ---------------------------------------------------------------------------
__global__ void softmax_kernel(float* __restrict__ att, int Tq, int Tn, int causal)
{
    long row = blockIdx.x;                  // over nbatch * Tq
    int t = (int)(row % Tq);
    float* a = att + row * (long)Tn;
    int limit = causal ? (t + 1) : Tn;
    int i = threadIdx.x;
    __shared__ float red[256];

    float v = (i < limit) ? a[i] : -INFINITY;
    red[i] = v; __syncthreads();
    for (int s = 128; s > 0; s >>= 1) {
        if (i < s) red[i] = fmaxf(red[i], red[i + s]);
        __syncthreads();
    }
    float mx = red[0]; __syncthreads();

    float e = (i < limit) ? __expf(v - mx) : 0.f;
    red[i] = e; __syncthreads();
    for (int s = 128; s > 0; s >>= 1) {
        if (i < s) red[i] += red[i + s];
        __syncthreads();
    }
    float inv = 1.f / red[0];
    if (i < Tn) a[i] = e * inv;
}

// ---------------------------------------------------------------------------
// LayerNorm: y = (x - mean) * rsqrt(var + eps) * w + b   (one block per row)
// ---------------------------------------------------------------------------
__global__ void layernorm_kernel(const float* __restrict__ x, float* __restrict__ y,
                                 const float* __restrict__ w, const float* __restrict__ b,
                                 int n)
{
    long row = blockIdx.x;
    x += row * (long)n; y += row * (long)n;
    int i = threadIdx.x;
    float s = 0.f, s2 = 0.f;
    for (int j = i; j < n; j += 256) { float v = x[j]; s += v; s2 += v * v; }
    __shared__ float r1[256], r2[256];
    r1[i] = s; r2[i] = s2; __syncthreads();
    for (int st = 128; st > 0; st >>= 1) {
        if (i < st) { r1[i] += r1[i + st]; r2[i] += r2[i + st]; }
        __syncthreads();
    }
    float mean = r1[0] / n;
    float var  = r2[0] / n - mean * mean;
    float rinv = rsqrtf(var + 1e-5f);
    for (int j = i; j < n; j += 256)
        y[j] = (x[j] - mean) * rinv * w[j] + b[j];
}

// ---------------------------------------------------------------------------
// Host-side helpers
// ---------------------------------------------------------------------------
static inline void gemm(hipStream_t s, const float* A, const float* Bm, const float* bias,
                        const float* res, float* C, int M, int N, int K,
                        int lda, int ldb, int ldc,
                        int nbz, int nbh, long sAb, long sAh, long sBb, long sBh,
                        long sCb, long sCh, float alpha, int flags)
{
    dim3 g((N + BN - 1) / BN, (M + BM - 1) / BM, nbz * nbh);
    gemm_bf16_wmma<<<g, 256, 0, s>>>(A, Bm, bias, res, C, M, N, K, lda, ldb, ldc,
                                     nbh, sAb, sAh, sBb, sBh, sCb, sCh, alpha, flags);
}

// Full multi-head attention: out = (softmax(qx Wq (kx Wk)^T / 8) (kx Wv)) Wo (+resid)
static void run_mha(hipStream_t s, const float* qx, int Tq, const float* kx, int Tn,
                    const float* wq, const float* wk, const float* wv, const float* wo,
                    int causal, const float* resid, float* out,
                    float* qb, float* kb, float* vb, float* att, float* ao)
{
    gemm(s, qx, wq, nullptr, nullptr, qb, B_ * Tq, H_, H_, H_, H_, H_,
         1, 1, 0, 0, 0, 0, 0, 0, 1.f, 0);
    gemm(s, kx, wk, nullptr, nullptr, kb, B_ * Tn, H_, H_, H_, H_, H_,
         1, 1, 0, 0, 0, 0, 0, 0, 1.f, 0);
    gemm(s, kx, wv, nullptr, nullptr, vb, B_ * Tn, H_, H_, H_, H_, H_,
         1, 1, 0, 0, 0, 0, 0, 0, 1.f, 0);
    // scores[b,h] = Q[b,h] K[b,h]^T / sqrt(HD)
    gemm(s, qb, kb, nullptr, nullptr, att, Tq, Tn, HD_, H_, H_, Tn,
         B_, NH_, (long)Tq * H_, HD_, (long)Tn * H_, HD_,
         (long)NH_ * Tq * Tn, (long)Tq * Tn, 0.125f, 1 /*B^T*/);
    softmax_kernel<<<B_ * NH_ * Tq, 256, 0, s>>>(att, Tq, Tn, causal);
    // context[b,h] = P[b,h] V[b,h]
    gemm(s, att, vb, nullptr, nullptr, ao, Tq, HD_, Tn, Tn, H_, H_,
         B_, NH_, (long)NH_ * Tq * Tn, (long)Tq * Tn, (long)Tn * H_, HD_,
         (long)Tq * H_, HD_, 1.f, 0);
    // output projection (+residual)
    gemm(s, ao, wo, nullptr, resid, out, B_ * Tq, H_, H_, H_, H_, H_,
         1, 1, 0, 0, 0, 0, 0, 0, 1.f, resid ? 4 : 0);
}

// ---------------------------------------------------------------------------
// Orchestration
// ---------------------------------------------------------------------------
extern "C" void kernel_launch(void* const* d_in, const int* in_sizes, int n_in,
                              void* d_out, int out_size, void* d_ws, size_t ws_size,
                              hipStream_t stream)
{
    const float* vfeat  = (const float*)d_in[0];   // [16,49,2048]
    const int*   tokens = (const int*)  d_in[1];   // [16,256]
    const int*   kwtok  = (const int*)  d_in[2];   // [16,64]
    const float* We     = (const float*)d_in[3];   // [V,H]
    const float* wpe    = (const float*)d_in[4];   // [512,H]
    const float* vf_w   = (const float*)d_in[5];   // [E,H]
    const float* vf_b   = (const float*)d_in[6];   // [H]
    const float* le_q   = (const float*)d_in[7];
    const float* le_k   = (const float*)d_in[8];
    const float* le_v   = (const float*)d_in[9];
    const float* le_o   = (const float*)d_in[10];
    const float* f_q    = (const float*)d_in[11];  // [L,H,H]
    const float* f_k    = (const float*)d_in[12];
    const float* f_v    = (const float*)d_in[13];
    const float* f_o    = (const float*)d_in[14];
    const float* f_fc   = (const float*)d_in[15];  // [L,H,F]
    const float* f_pr   = (const float*)d_in[16];  // [L,F,H]
    const float* f_ln1w = (const float*)d_in[17];
    const float* f_ln1b = (const float*)d_in[18];
    const float* f_ln2w = (const float*)d_in[19];
    const float* f_ln2b = (const float*)d_in[20];
    const float* d_sq   = (const float*)d_in[21];
    const float* d_sk   = (const float*)d_in[22];
    const float* d_sv   = (const float*)d_in[23];
    const float* d_so   = (const float*)d_in[24];
    const float* d_cq   = (const float*)d_in[25];
    const float* d_ck   = (const float*)d_in[26];
    const float* d_cv   = (const float*)d_in[27];
    const float* d_co   = (const float*)d_in[28];
    const float* d_fc   = (const float*)d_in[29];
    const float* d_pr   = (const float*)d_in[30];
    const float* d_ln1w = (const float*)d_in[31];
    const float* d_ln1b = (const float*)d_in[32];
    const float* d_ln2w = (const float*)d_in[33];
    const float* d_ln2b = (const float*)d_in[34];
    const float* d_ln3w = (const float*)d_in[35];
    const float* d_ln3b = (const float*)d_in[36];
    (void)in_sizes; (void)n_in; (void)out_size; (void)ws_size;

    // workspace carve-out (floats)
    float* ws = (float*)d_ws;
    size_t off = 0;
    auto alloc = [&](size_t n) { float* p = ws + off; off += n; return p; };
    float* ke  = alloc((size_t)B_ * KW_ * H_);       // keyword embeddings / encoder out
    float* x   = alloc((size_t)B_ * T_ * H_);        // decoder stream
    float* enc = alloc((size_t)B_ * VT_ * H_);       // fused visual stream
    float* qb  = alloc((size_t)B_ * T_ * H_);
    float* kb  = alloc((size_t)B_ * T_ * H_);
    float* vb  = alloc((size_t)B_ * T_ * H_);
    float* att = alloc((size_t)B_ * NH_ * T_ * T_);
    float* ao  = alloc((size_t)B_ * T_ * H_);
    float* h1  = alloc((size_t)B_ * T_ * F_);
    float* t2  = alloc((size_t)B_ * T_ * H_);

    // ---- keyword pathway: ke = MHA(We[kw]) (no residual, no LN) ----
    embed_kernel<<<B_ * KW_, 256, 0, stream>>>(kwtok, We, nullptr, ke, KW_, H_);
    run_mha(stream, ke, KW_, ke, KW_, le_q, le_k, le_v, le_o, 0, nullptr, ke,
            qb, kb, vb, att, ao);

    // ---- token + position embedding ----
    embed_kernel<<<B_ * T_, 256, 0, stream>>>(tokens, We, wpe, x, T_, H_);

    // ---- visual projection: enc0 = vfeat @ vf_proj_w + b ----
    gemm(stream, vfeat, vf_w, vf_b, nullptr, enc, B_ * VT_, H_, E_, E_, H_, H_,
         1, 1, 0, 0, 0, 0, 0, 0, 1.f, 2 /*bias*/);

    for (int i = 0; i < L_; ++i) {
        const long hh = (long)i * H_ * H_;
        const long hf = (long)i * H_ * F_;
        const long lh = (long)i * H_;

        // ---- fuser: cross-attn (vf -> keywords), LN1 ----
        run_mha(stream, enc, VT_, ke, KW_, f_q + hh, f_k + hh, f_v + hh, f_o + hh,
                0, enc, t2, qb, kb, vb, att, ao);
        layernorm_kernel<<<B_ * VT_, 256, 0, stream>>>(t2, enc, f_ln1w + lh, f_ln1b + lh, H_);
        // ---- fuser MLP + LN2 ----
        gemm(stream, enc, f_fc + hf, nullptr, nullptr, h1, B_ * VT_, F_, H_, H_, F_, F_,
             1, 1, 0, 0, 0, 0, 0, 0, 1.f, 8 /*gelu*/);
        gemm(stream, h1, f_pr + hf, nullptr, enc, t2, B_ * VT_, H_, F_, F_, H_, H_,
             1, 1, 0, 0, 0, 0, 0, 0, 1.f, 4 /*res*/);
        layernorm_kernel<<<B_ * VT_, 256, 0, stream>>>(t2, enc, f_ln2w + lh, f_ln2b + lh, H_);

        // ---- decoder: causal self-attn + LN1 ----
        run_mha(stream, x, T_, x, T_, d_sq + hh, d_sk + hh, d_sv + hh, d_so + hh,
                1, x, t2, qb, kb, vb, att, ao);
        layernorm_kernel<<<B_ * T_, 256, 0, stream>>>(t2, x, d_ln1w + lh, d_ln1b + lh, H_);
        // ---- decoder: cross-attn to enc + LN2 ----
        run_mha(stream, x, T_, enc, VT_, d_cq + hh, d_ck + hh, d_cv + hh, d_co + hh,
                0, x, t2, qb, kb, vb, att, ao);
        layernorm_kernel<<<B_ * T_, 256, 0, stream>>>(t2, x, d_ln2w + lh, d_ln2b + lh, H_);
        // ---- decoder MLP + LN3 ----
        gemm(stream, x, d_fc + hf, nullptr, nullptr, h1, B_ * T_, F_, H_, H_, F_, F_,
             1, 1, 0, 0, 0, 0, 0, 0, 1.f, 8);
        gemm(stream, h1, d_pr + hf, nullptr, x, t2, B_ * T_, H_, F_, F_, H_, H_,
             1, 1, 0, 0, 0, 0, 0, 0, 1.f, 4);
        layernorm_kernel<<<B_ * T_, 256, 0, stream>>>(t2, x, d_ln3w + lh, d_ln3b + lh, H_);
    }

    // ---- tied lm_head: logits = x @ We^T ----
    gemm(stream, x, We, nullptr, nullptr, (float*)d_out, B_ * T_, V_, H_, H_, H_, V_,
         1, 1, 0, 0, 0, 0, 0, 0, 1.f, 1 /*B^T*/);
}
